// MultiStageDetector_52312701666057
// MI455X (gfx1250) — matrix operand order repaired and checked by reading, not compiled, not run on hardware
//
#include <hip/hip_runtime.h>
#include <math.h>

// ---------------------------------------------------------------------------
// Types for CDNA5 WMMA (wave32)
// ---------------------------------------------------------------------------
typedef __bf16 bf16;
typedef __bf16 v16bf __attribute__((ext_vector_type(16)));
typedef float  v8f   __attribute__((ext_vector_type(8)));

// ---------------------------------------------------------------------------
// Tensor Data Mover (TDM) support: async 2D tile copy global(bf16) -> LDS,
// tracked by TENSORcnt. Descriptor packed per CDNA5 ISA ch.8 (D# groups).
// Guarded so compilation never depends on the builtin being present.
// ---------------------------------------------------------------------------
#if defined(__AMDGCN__) && __has_builtin(__builtin_amdgcn_tensor_load_to_lds) && \
    __has_builtin(__builtin_amdgcn_s_wait_tensorcnt)
#define HAVE_TDM 1
#else
#define HAVE_TDM 0
#endif

typedef unsigned int u32x4 __attribute__((ext_vector_type(4)));
typedef int          i32x4 __attribute__((ext_vector_type(4)));
typedef int          i32x8 __attribute__((ext_vector_type(8)));

// Issue one TDM tile load: rows x cols bf16 elements, global row stride
// row_stride_elems, into LDS byte offset lds_off (rows packed contiguously).
__device__ __forceinline__ void tdm_load_tile(const bf16* gsrc, unsigned lds_off,
                                              int rows, int cols,
                                              int row_stride_elems) {
#if HAVE_TDM
  const unsigned long long ga = (unsigned long long)(uintptr_t)gsrc;
  // ---- D# group 0: count=1 | lds_addr | global_addr | type=2 ----
  const unsigned long long g0q0 = 1ull | ((unsigned long long)lds_off << 32);
  const unsigned long long g0q1 = (ga & 0x01FFFFFFFFFFFFFFull) | (2ull << 62);
  u32x4 g0 = {(unsigned)g0q0, (unsigned)(g0q0 >> 32),
              (unsigned)g0q1, (unsigned)(g0q1 >> 32)};
  // ---- D# group 1 ----
  const unsigned long long td0 = (unsigned)cols;            // tensor_dim0
  const unsigned long long td1 = (unsigned)rows;            // tensor_dim1
  const unsigned long long s0  = (unsigned long long)row_stride_elems;
  // bits: [17:16]=data_size(1=2B), [79:48]=tensor_dim0, [111:80]=tensor_dim1,
  //       [127:112]=tile_dim0, [143:128]=tile_dim1, [207:160]=dim0_stride
  const unsigned long long q0 = (1ull << 16) | ((td0 & 0xFFFFull) << 48);
  const unsigned long long q1 = (td0 >> 16) | (td1 << 16) |
                                ((td0 & 0xFFFFull) << 48);
  const unsigned long long q2 = (td1 & 0xFFFFull) | ((s0 & 0xFFFFFFFFull) << 32);
  const unsigned long long q3 = (s0 >> 32);
  i32x8 g1 = {(int)(unsigned)q0, (int)(unsigned)(q0 >> 32),
              (int)(unsigned)q1, (int)(unsigned)(q1 >> 32),
              (int)(unsigned)q2, (int)(unsigned)(q2 >> 32),
              (int)(unsigned)q3, (int)(unsigned)(q3 >> 32)};
  i32x4 z4 = {0, 0, 0, 0};
#if defined(__clang_major__) && __clang_major__ >= 23
  i32x8 z8 = {0, 0, 0, 0, 0, 0, 0, 0};
  __builtin_amdgcn_tensor_load_to_lds(g0, g1, z4, z4, z8, 0);
#else
  __builtin_amdgcn_tensor_load_to_lds(g0, g1, z4, z4, 0);
#endif
#else
  (void)gsrc; (void)lds_off; (void)rows; (void)cols; (void)row_stride_elems;
#endif
}

__device__ __forceinline__ void tdm_wait() {
#if HAVE_TDM
  __builtin_amdgcn_s_wait_tensorcnt((short)0);
#endif
}

// ---------------------------------------------------------------------------
// NT-GEMM with WMMA bf16:  C[M,N] = act( A[M,K] * B[N,K]^T + bias )
//   Tile: 128(M) x 64(N) x 32(K); 256 threads = 8 waves; each wave owns a
//   32x32 sub-tile -> 4 x v_wmma_f32_16x16x32_bf16 per K-step.
//   bf16 operands are staged by TDM (tensor_load_to_lds + s_wait_tensorcnt);
//   f32 operands use vectorized load+convert staging. Edge tiles take a
//   guarded scalar path. biasMode: 0=none, 1=per-row (convs), 2=per-col (FC).
// ---------------------------------------------------------------------------
template <typename TA, typename TB>
__global__ void gemm_nt_wmma(const TA* __restrict__ Ag, long strideA, int lda,
                             const TB* __restrict__ Bg, long strideB, int ldb,
                             bf16* __restrict__ Cg, long strideC,
                             const float* __restrict__ bias, int biasMode,
                             int relu, int M, int N, int K) {
  __shared__ bf16 As[128][32];
  __shared__ bf16 Bs[64][32];

  const TA* A = Ag + (long)blockIdx.z * strideA;
  const TB* B = Bg + (long)blockIdx.z * strideB;
  bf16*     C = Cg + (long)blockIdx.z * strideC;

  const int m0 = blockIdx.y * 128;
  const int n0 = blockIdx.x * 64;
  const int tid  = threadIdx.x;
  const int lane = tid & 31;
  const int wave = tid >> 5;
  const int wm   = wave & 3;   // 4 M-groups of 32 rows
  const int wn   = wave >> 2;  // 2 N-groups of 32 cols
  const int half = lane >> 4;  // 0: lanes 0-15, 1: lanes 16-31
  const int r    = lane & 15;

  // staging coordinates
  const int arow = tid >> 1;         // 0..127, 16 k-elems per thread
  const int acol = (tid & 1) * 16;   // 0 / 16
  const int brow = tid >> 2;         // 0..63, 8 k-elems per thread
  const int bcol = (tid & 3) * 8;    // 0,8,16,24

  const unsigned ldsAsOff = (unsigned)(uintptr_t)(&As[0][0]);
  const unsigned ldsBsOff = (unsigned)(uintptr_t)(&Bs[0][0]);

  const bool fullMN = (m0 + 128 <= M) && (n0 + 64 <= N);

  v8f c00 = {}, c01 = {}, c10 = {}, c11 = {};

  for (int kk = 0; kk < K; kk += 32) {
    const bool fullK = (kk + 32 <= K);
    if (fullMN && fullK) {
      bool usedTdm = false;
      // ---------------- A staging ----------------
      if (HAVE_TDM && sizeof(TA) == 2) {
        if (wave == 0)
          tdm_load_tile((const bf16*)(const void*)(A + (long)m0 * lda + kk),
                        ldsAsOff, 128, 32, lda);
        usedTdm = true;
      } else {
        const TA* pa = A + (long)(m0 + arow) * lda + kk + acol;
        union { bf16 h[16]; uint4 v[2]; } ua;
#pragma unroll
        for (int j = 0; j < 16; ++j) ua.h[j] = (bf16)(float)pa[j];
        uint4* dst = reinterpret_cast<uint4*>(&As[arow][acol]);
        dst[0] = ua.v[0];
        dst[1] = ua.v[1];
        if (kk + 32 < K)
          __builtin_prefetch(pa + 32, 0, 0);
      }
      // ---------------- B staging ----------------
      if (HAVE_TDM && sizeof(TB) == 2) {
        if (wave == 0)
          tdm_load_tile((const bf16*)(const void*)(B + (long)n0 * ldb + kk),
                        ldsBsOff, 64, 32, ldb);
        usedTdm = true;
      } else {
        const TB* pb = B + (long)(n0 + brow) * ldb + kk + bcol;
        union { bf16 h[8]; uint4 v; } ub;
#pragma unroll
        for (int j = 0; j < 8; ++j) ub.h[j] = (bf16)(float)pb[j];
        *reinterpret_cast<uint4*>(&Bs[brow][bcol]) = ub.v;
        if (kk + 32 < K)
          __builtin_prefetch(pb + 32, 0, 0);
      }
      if (usedTdm && wave == 0) tdm_wait();
    } else {
      // ---------------- guarded scalar staging (edge tiles) ----------------
      const int m = m0 + arow;
#pragma unroll
      for (int j = 0; j < 16; ++j) {
        const int k = kk + acol + j;
        float v = (m < M && k < K) ? (float)A[(long)m * lda + k] : 0.0f;
        As[arow][acol + j] = (bf16)v;
      }
      const int n = n0 + brow;
#pragma unroll
      for (int j = 0; j < 8; ++j) {
        const int k = kk + bcol + j;
        float v = (n < N && k < K) ? (float)B[(long)n * ldb + k] : 0.0f;
        Bs[brow][bcol + j] = (bf16)v;
      }
    }
    __syncthreads();

    // ---- fragments per ISA 16-bit layouts ----
    // A 16x32: lanes<16 hold K {0..7,16..23}, lanes>=16 hold K {8..15,24..31}
    // B 32x16: lanes<16 hold K {0..7,16..23}, lanes>=16 hold K {8..15,24..31} (col = lane%16)
    v16bf a0, a1, b0, b1;
#pragma unroll
    for (int i = 0; i < 16; ++i) {
      const int ka = ((i < 8) ? i : i + 8) + half * 8;
      const int kb = ka;
      a0[i] = As[wm * 32 + r][ka];
      a1[i] = As[wm * 32 + 16 + r][ka];
      b0[i] = Bs[wn * 32 + r][kb];
      b1[i] = Bs[wn * 32 + 16 + r][kb];
    }
#if defined(__HIP_DEVICE_COMPILE__)
    c00 = __builtin_amdgcn_wmma_f32_16x16x32_bf16(false, a0, false, b0,
                                                  (short)0, c00, false, false);
    c01 = __builtin_amdgcn_wmma_f32_16x16x32_bf16(false, a0, false, b1,
                                                  (short)0, c01, false, false);
    c10 = __builtin_amdgcn_wmma_f32_16x16x32_bf16(false, a1, false, b0,
                                                  (short)0, c10, false, false);
    c11 = __builtin_amdgcn_wmma_f32_16x16x32_bf16(false, a1, false, b1,
                                                  (short)0, c11, false, false);
#endif
    __syncthreads();
  }

  // ---- epilogue: C frag layout = VGPR q -> M = q + 8*half, N = lane%16 ----
#pragma unroll
  for (int tm = 0; tm < 2; ++tm) {
#pragma unroll
    for (int tn = 0; tn < 2; ++tn) {
      v8f cc = (tm == 0) ? (tn == 0 ? c00 : c01) : (tn == 0 ? c10 : c11);
#pragma unroll
      for (int q = 0; q < 8; ++q) {
        const int m = m0 + wm * 32 + tm * 16 + half * 8 + q;
        const int n = n0 + wn * 32 + tn * 16 + r;
        float v = cc[q];
        if (biasMode == 1)      v += (m < M) ? bias[m] : 0.0f;
        else if (biasMode == 2) v += (n < N) ? bias[n] : 0.0f;
        if (relu) v = fmaxf(v, 0.0f);
        if (m < M && n < N) C[(long)m * N + n] = (bf16)v;
      }
    }
  }
}

// ---------------------------------------------------------------------------
// im2col for 3x3 pad=1 conv, producing transposed patches [B][H*W][Kpad] bf16
// (NT layout: GEMM B-operand with K contiguous). k >= Ci*9 zero-filled.
// ---------------------------------------------------------------------------
template <typename T>
__global__ void im2col3x3(const T* __restrict__ in, bf16* __restrict__ out,
                          int B, int Ci, int H, int W, int Kpad) {
  long idx = (long)blockIdx.x * blockDim.x + threadIdx.x;
  long total = (long)B * H * W * Kpad;
  if (idx >= total) return;
  int k = (int)(idx % Kpad);
  long rest = idx / Kpad;
  int x = (int)(rest % W); rest /= W;
  int y = (int)(rest % H); rest /= H;
  int b = (int)rest;
  float v = 0.0f;
  const int Kact = Ci * 9;
  if (k < Kact) {
    int c = k / 9, ky = (k % 9) / 3, kx = k % 3;
    int yy = y + ky - 1, xx = x + kx - 1;
    if (yy >= 0 && yy < H && xx >= 0 && xx < W)
      v = (float)in[(((long)b * Ci + c) * H + yy) * W + xx];
  }
  out[idx] = (bf16)v;
}

// ---------------------------------------------------------------------------
// 2x2 max pool (stride 2), bf16
// ---------------------------------------------------------------------------
__global__ void maxpool2(const bf16* __restrict__ in, bf16* __restrict__ out,
                         int B, int C, int H, int W) {
  const int Ho = H / 2, Wo = W / 2;
  long idx = (long)blockIdx.x * blockDim.x + threadIdx.x;
  long total = (long)B * C * Ho * Wo;
  if (idx >= total) return;
  int x = (int)(idx % Wo);
  long rest = idx / Wo;
  int y = (int)(rest % Ho); rest /= Ho;
  int c = (int)(rest % C);
  int b = (int)(rest / C);
  const bf16* p = in + (((long)b * C + c) * H + y * 2) * W + x * 2;
  float m = fmaxf(fmaxf((float)p[0], (float)p[1]),
                  fmaxf((float)p[W], (float)p[W + 1]));
  out[idx] = (bf16)m;
}

// ---------------------------------------------------------------------------
// RPN 1x1 heads: deltas[b][hw][0..3] from boxw, scores[b][hw] = cls channel 1
// r2 layout: [B][256][HW] bf16
// ---------------------------------------------------------------------------
__global__ void rpn_head(const bf16* __restrict__ r2,
                         const float* __restrict__ boxw, const float* __restrict__ boxb,
                         const float* __restrict__ clsw, const float* __restrict__ clsb,
                         float* __restrict__ deltas, float* __restrict__ scores,
                         int HW) {
  int idx = blockIdx.x * blockDim.x + threadIdx.x;
  int total = 2 * HW * 5;
  if (idx >= total) return;
  int j  = idx % 5;
  int hw = (idx / 5) % HW;
  int b  = idx / (5 * HW);
  const bf16* base = r2 + (long)b * 256 * HW + hw;
  const float* w;
  float acc;
  if (j < 4) { w = boxw + j * 256; acc = boxb[j]; }
  else       { w = clsw + 256;     acc = clsb[1]; }
  for (int c = 0; c < 256; ++c) acc += (float)base[(long)c * HW] * w[c];
  if (j < 4) deltas[((long)b * HW + hw) * 4 + j] = acc;
  else       scores[(long)b * HW + hw] = acc;
}

// ---------------------------------------------------------------------------
// Anchor decode (ANCHOR=4, clip to [0,Wf]/[0,Hf])
// ---------------------------------------------------------------------------
__global__ void decode_boxes(const float* __restrict__ deltas,
                             float* __restrict__ boxes, int Hf, int Wf) {
  int idx = blockIdx.x * blockDim.x + threadIdx.x;
  int HW = Hf * Wf;
  if (idx >= 2 * HW) return;
  int hw = idx % HW;
  float cx = (float)(hw % Wf) + 0.5f;
  float cy = (float)(hw / Wf) + 0.5f;
  const float* d = deltas + (long)idx * 4;
  float px = cx + d[0] * 4.0f;
  float py = cy + d[1] * 4.0f;
  float pw = 4.0f * expf(fminf(fmaxf(d[2], -4.0f), 4.0f));
  float ph = 4.0f * expf(fminf(fmaxf(d[3], -4.0f), 4.0f));
  float* o = boxes + (long)idx * 4;
  o[0] = fminf(fmaxf(px - pw * 0.5f, 0.0f), (float)Wf);
  o[1] = fminf(fmaxf(py - ph * 0.5f, 0.0f), (float)Hf);
  o[2] = fminf(fmaxf(px + pw * 0.5f, 0.0f), (float)Wf);
  o[3] = fminf(fmaxf(py + ph * 0.5f, 0.0f), (float)Hf);
}

// ---------------------------------------------------------------------------
// NMS: one block per image, 784 boxes resident in LDS. Stable descending sort
// via ranking, greedy suppression, then top-100 (kept first, then suppressed
// in sorted order to mimic top_k tie-breaking over -inf).
// ---------------------------------------------------------------------------
#define NMS_N 784
__global__ void nms_kernel(const float* __restrict__ boxes,
                           const float* __restrict__ scores,
                           float* __restrict__ rois) {
  __shared__ float bx[NMS_N * 4];
  __shared__ float sc[NMS_N];
  __shared__ int   order[NMS_N];
  __shared__ unsigned char keep[NMS_N];
  __shared__ int   sel[100];

  const int b = blockIdx.x, t = threadIdx.x, nt = blockDim.x;
  const float* bb = boxes  + (long)b * NMS_N * 4;
  const float* ss = scores + (long)b * NMS_N;

  for (int i = t; i < NMS_N; i += nt) {
    for (int q = 0; q < 4; ++q) bx[i * 4 + q] = bb[i * 4 + q];
    sc[i] = ss[i];
    keep[i] = 1;
  }
  __syncthreads();

  // stable rank for descending sort (matches argsort(-scores))
  for (int i = t; i < NMS_N; i += nt) {
    const float si = sc[i];
    int rank = 0;
    for (int j = 0; j < NMS_N; ++j) {
      const float sj = sc[j];
      rank += (sj > si) || (sj == si && j < i);
    }
    order[rank] = i;
  }
  __syncthreads();

  for (int i = 0; i < NMS_N - 1; ++i) {
    if (keep[i]) {
      const int oi = order[i];
      const float x1 = bx[oi * 4], y1 = bx[oi * 4 + 1];
      const float x2 = bx[oi * 4 + 2], y2 = bx[oi * 4 + 3];
      const float ai = (x2 - x1) * (y2 - y1);
      for (int j = i + 1 + t; j < NMS_N; j += nt) {
        if (!keep[j]) continue;
        const int oj = order[j];
        const float u1 = bx[oj * 4], v1 = bx[oj * 4 + 1];
        const float u2 = bx[oj * 4 + 2], v2 = bx[oj * 4 + 3];
        const float aj = (u2 - u1) * (v2 - v1);
        const float ix1 = fmaxf(x1, u1), iy1 = fmaxf(y1, v1);
        const float ix2 = fminf(x2, u2), iy2 = fminf(y2, v2);
        const float iw = fmaxf(ix2 - ix1, 0.0f), ih = fmaxf(iy2 - iy1, 0.0f);
        const float inter = iw * ih;
        const float iou = inter / (ai + aj - inter + 1e-6f);
        if (iou > 0.5f) keep[j] = 0;
      }
    }
    __syncthreads();
  }

  if (t == 0) {
    int cnt = 0;
    for (int i = 0; i < NMS_N && cnt < 100; ++i) if (keep[i])  sel[cnt++] = order[i];
    for (int i = 0; i < NMS_N && cnt < 100; ++i) if (!keep[i]) sel[cnt++] = order[i];
  }
  __syncthreads();

  for (int r = t; r < 100; r += nt) {
    const int o = sel[r];
    for (int q = 0; q < 4; ++q) rois[((long)b * 100 + r) * 4 + q] = bx[o * 4 + q];
  }
}

// ---------------------------------------------------------------------------
// ROI max-pool: thread per (b, roi, c); 14x14 samples -> 7x7 (2x2 max)
// pooled row layout: [c][7][7], row length C*49 = 50176, row = b*100 + roi
// ---------------------------------------------------------------------------
__global__ void roi_pool(const bf16* __restrict__ feat,
                         const float* __restrict__ rois,
                         bf16* __restrict__ pooled, int C, int Hf, int Wf) {
  long idx = (long)blockIdx.x * blockDim.x + threadIdx.x;
  long total = (long)2 * 100 * C;
  if (idx >= total) return;
  int c = (int)(idx % C);
  int r = (int)((idx / C) % 100);
  int b = (int)(idx / ((long)C * 100));
  const float* box = rois + ((long)b * 100 + r) * 4;
  const float x1 = box[0], y1 = box[1], x2 = box[2], y2 = box[3];
  int xi[14], yi[14];
  for (int g = 0; g < 14; ++g) {
    float xs = x1 + (g + 0.5f) * (x2 - x1) * (1.0f / 14.0f);
    float ys = y1 + (g + 0.5f) * (y2 - y1) * (1.0f / 14.0f);
    xi[g] = min(max((int)floorf(xs), 0), Wf - 1);
    yi[g] = min(max((int)floorf(ys), 0), Hf - 1);
  }
  const bf16* f = feat + ((long)b * C + c) * Hf * Wf;
  bf16* o = pooled + ((long)b * 100 + r) * ((long)C * 49) + (long)c * 49;
  for (int py = 0; py < 7; ++py)
    for (int px = 0; px < 7; ++px) {
      float m = -1e30f;
      for (int sy = 0; sy < 2; ++sy)
        for (int sx = 0; sx < 2; ++sx)
          m = fmaxf(m, (float)f[yi[py * 2 + sy] * Wf + xi[px * 2 + sx]]);
      o[py * 7 + px] = (bf16)m;
    }
}

// ---------------------------------------------------------------------------
// Final 6-wide head: out[200][6] f32 = h2 @ fw3^T + fb3 (no activation)
// ---------------------------------------------------------------------------
__global__ void fc3_kernel(const bf16* __restrict__ h2,
                           const float* __restrict__ w,
                           const float* __restrict__ bias,
                           float* __restrict__ out) {
  int idx = blockIdx.x * blockDim.x + threadIdx.x;
  if (idx >= 200 * 6) return;
  const int j = idx % 6, row = idx / 6;
  float acc = bias[j];
  const bf16*  hp = h2 + (long)row * 4096;
  const float* wp = w + (long)j * 4096;
  for (int k = 0; k < 4096; ++k) acc += (float)hp[k] * wp[k];
  out[idx] = acc;
}

// ---------------------------------------------------------------------------
// Host-side layer drivers
// ---------------------------------------------------------------------------
static void run_conv_f32in(hipStream_t s, const float* in, const float* W,
                           const float* bias, bf16* out, bf16* col,
                           int B, int Ci, int H, int Wd, int Co) {
  const int HW = H * Wd, Kact = Ci * 9, Kpad = ((Kact + 31) / 32) * 32;
  long total = (long)B * HW * Kpad;
  im2col3x3<float><<<dim3((unsigned)((total + 255) / 256)), 256, 0, s>>>(
      in, col, B, Ci, H, Wd, Kpad);
  dim3 g((HW + 63) / 64, (Co + 127) / 128, B);
  gemm_nt_wmma<float, bf16><<<g, 256, 0, s>>>(
      W, 0L, Kact, col, (long)HW * Kpad, Kpad, out, (long)Co * HW,
      bias, /*biasMode=*/1, /*relu=*/1, Co, HW, Kact);
}

static void run_conv_bf16in(hipStream_t s, const bf16* in, const float* W,
                            const float* bias, bf16* out, bf16* col,
                            int B, int Ci, int H, int Wd, int Co) {
  const int HW = H * Wd, Kact = Ci * 9, Kpad = ((Kact + 31) / 32) * 32;
  long total = (long)B * HW * Kpad;
  im2col3x3<bf16><<<dim3((unsigned)((total + 255) / 256)), 256, 0, s>>>(
      in, col, B, Ci, H, Wd, Kpad);
  dim3 g((HW + 63) / 64, (Co + 127) / 128, B);
  gemm_nt_wmma<float, bf16><<<g, 256, 0, s>>>(
      W, 0L, Kact, col, (long)HW * Kpad, Kpad, out, (long)Co * HW,
      bias, /*biasMode=*/1, /*relu=*/1, Co, HW, Kact);
}

static void run_pool(hipStream_t s, const bf16* in, bf16* out,
                     int B, int C, int H, int W) {
  long total = (long)B * C * (H / 2) * (W / 2);
  maxpool2<<<dim3((unsigned)((total + 255) / 256)), 256, 0, s>>>(in, out, B, C, H, W);
}

extern "C" void kernel_launch(void* const* d_in, const int* in_sizes, int n_in,
                              void* d_out, int out_size, void* d_ws, size_t ws_size,
                              hipStream_t stream) {
  (void)in_sizes; (void)n_in; (void)out_size; (void)ws_size;
  const float* x = (const float*)d_in[0];
  const float *bw[9], *bb[9];
  for (int i = 0; i < 9; ++i) {
    bw[i] = (const float*)d_in[1 + 2 * i];
    bb[i] = (const float*)d_in[2 + 2 * i];
  }
  const float* rw1  = (const float*)d_in[19];
  const float* rb1  = (const float*)d_in[20];
  const float* rw2  = (const float*)d_in[21];
  const float* rb2  = (const float*)d_in[22];
  const float* boxw = (const float*)d_in[23];
  const float* boxb = (const float*)d_in[24];
  const float* clsw = (const float*)d_in[25];
  const float* clsb = (const float*)d_in[26];
  const float* fw1  = (const float*)d_in[27];
  const float* fb1  = (const float*)d_in[28];
  const float* fw2  = (const float*)d_in[29];
  const float* fb2  = (const float*)d_in[30];
  const float* fw3  = (const float*)d_in[31];
  const float* fb3  = (const float*)d_in[32];
  float* out = (float*)d_out;

  // -------- workspace bump allocator --------
  char* ws = (char*)d_ws;
  size_t off = 0;
  auto alloc = [&](size_t bytes) -> void* {
    void* p = ws + off;
    off += (bytes + 255) & ~(size_t)255;
    return p;
  };
  const int B = 2;
  bf16* col  = (bf16*)alloc((size_t)14450688 * 2);            // max im2col
  bf16* a1   = (bf16*)alloc((size_t)B * 64  * 224 * 224 * 2);
  bf16* p1   = (bf16*)alloc((size_t)B * 64  * 112 * 112 * 2);
  bf16* a2   = (bf16*)alloc((size_t)B * 128 * 112 * 112 * 2);
  bf16* p2   = (bf16*)alloc((size_t)B * 128 * 56 * 56 * 2);
  bf16* a3   = (bf16*)alloc((size_t)B * 256 * 56 * 56 * 2);
  bf16* a4   = (bf16*)alloc((size_t)B * 256 * 56 * 56 * 2);
  bf16* a5   = (bf16*)alloc((size_t)B * 512 * 56 * 56 * 2);
  bf16* p5   = (bf16*)alloc((size_t)B * 512 * 28 * 28 * 2);
  bf16* a6   = (bf16*)alloc((size_t)B * 512 * 28 * 28 * 2);
  bf16* a7   = (bf16*)alloc((size_t)B * 512 * 28 * 28 * 2);
  bf16* a8   = (bf16*)alloc((size_t)B * 1024 * 28 * 28 * 2);
  bf16* feat = (bf16*)alloc((size_t)B * 1024 * 28 * 28 * 2);
  bf16* r1   = (bf16*)alloc((size_t)B * 256 * 28 * 28 * 2);
  bf16* r2   = (bf16*)alloc((size_t)B * 256 * 28 * 28 * 2);
  bf16* pooled = (bf16*)alloc((size_t)200 * 50176 * 2);
  bf16* h1   = (bf16*)alloc((size_t)200 * 4096 * 2);
  bf16* h2   = (bf16*)alloc((size_t)200 * 4096 * 2);
  float* deltas = (float*)alloc((size_t)B * 784 * 4 * 4);
  float* scores = (float*)alloc((size_t)B * 784 * 4);
  float* dboxes = (float*)alloc((size_t)B * 784 * 4 * 4);
  float* rois   = (float*)alloc((size_t)B * 100 * 4 * 4);

  // -------- backbone --------
  run_conv_f32in (stream, x,   bw[0], bb[0], a1, col, B,   3, 224, 224,   64);
  run_pool       (stream, a1, p1, B,  64, 224, 224);
  run_conv_bf16in(stream, p1,  bw[1], bb[1], a2, col, B,  64, 112, 112,  128);
  run_pool       (stream, a2, p2, B, 128, 112, 112);
  run_conv_bf16in(stream, p2,  bw[2], bb[2], a3, col, B, 128,  56,  56,  256);
  run_conv_bf16in(stream, a3,  bw[3], bb[3], a4, col, B, 256,  56,  56,  256);
  run_conv_bf16in(stream, a4,  bw[4], bb[4], a5, col, B, 256,  56,  56,  512);
  run_pool       (stream, a5, p5, B, 512, 56, 56);
  run_conv_bf16in(stream, p5,  bw[5], bb[5], a6, col, B, 512,  28,  28,  512);
  run_conv_bf16in(stream, a6,  bw[6], bb[6], a7, col, B, 512,  28,  28,  512);
  run_conv_bf16in(stream, a7,  bw[7], bb[7], a8, col, B, 512,  28,  28, 1024);
  run_conv_bf16in(stream, a8,  bw[8], bb[8], feat, col, B, 1024, 28, 28, 1024);

  // -------- RPN --------
  run_conv_bf16in(stream, feat, rw1, rb1, r1, col, B, 1024, 28, 28, 256);
  run_conv_bf16in(stream, r1,   rw2, rb2, r2, col, B,  256, 28, 28, 256);
  {
    const int HW = 28 * 28;
    int total = 2 * HW * 5;
    rpn_head<<<(total + 255) / 256, 256, 0, stream>>>(r2, boxw, boxb, clsw, clsb,
                                                      deltas, scores, HW);
    decode_boxes<<<(2 * HW + 255) / 256, 256, 0, stream>>>(deltas, dboxes, 28, 28);
    nms_kernel<<<2, 256, 0, stream>>>(dboxes, scores, rois);
    long rp = (long)2 * 100 * 1024;
    roi_pool<<<(unsigned)((rp + 255) / 256), 256, 0, stream>>>(feat, rois, pooled,
                                                               1024, 28, 28);
  }

  // -------- FC head (WMMA GEMMs) --------
  {
    dim3 g1((4096 + 63) / 64, (200 + 127) / 128, 1);
    gemm_nt_wmma<bf16, float><<<g1, 256, 0, stream>>>(
        pooled, 0L, 50176, fw1, 0L, 50176, h1, 0L,
        fb1, /*biasMode=*/2, /*relu=*/1, 200, 4096, 50176);
    dim3 g2((4096 + 63) / 64, (200 + 127) / 128, 1);
    gemm_nt_wmma<bf16, float><<<g2, 256, 0, stream>>>(
        h1, 0L, 4096, fw2, 0L, 4096, h2, 0L,
        fb2, /*biasMode=*/2, /*relu=*/1, 200, 4096, 4096);
    fc3_kernel<<<(1200 + 255) / 256, 256, 0, stream>>>(h2, fw3, fb3, out);
  }
}
